// TemporalEncoder_88562225643866
// MI455X (gfx1250) — compile-verified, matrix operand order checked by reading, not run on hardware
//
#include <hip/hip_runtime.h>
#include <hip/hip_bf16.h>

// ---------------------------------------------------------------------------
// TemporalEncoder: 2-layer BiLSTM (B=16,T=1024,D=2048,H=512) + attn pooling.
//   Phase 1: input-gate GEMMs with v_wmma_f32_16x16x32_bf16 (f32 accumulate).
//   Phase 2: sequential scans with v_wmma_f32_16x16x4_f32, w_hh resident in
//            LDS, h staged per step via global_load_async_to_lds_b128,
//            grid-wide barrier per step (32 cooperating WGs, double-buffered h).
//   Phase 3: attention pooling (softmax over T, weighted sum).
// ---------------------------------------------------------------------------

typedef __attribute__((ext_vector_type(16))) __bf16 v16bf;
typedef __attribute__((ext_vector_type(8)))  __bf16 v8bf;
typedef __attribute__((ext_vector_type(4)))  __bf16 v4bf;
typedef __attribute__((ext_vector_type(8)))  float  v8f;
typedef __attribute__((ext_vector_type(2)))  float  v2f;

#define CB 16      // batch
#define CT 1024    // time
#define CH 512     // hidden
#define C4H 2048   // 4*H
#define C2H 1024   // 2*H
#define NWG 32     // workgroups cooperating in one scan (H/16)

// ------------------ async global -> LDS copy (CDNA5, ASYNCcnt) --------------
__device__ __forceinline__ void async_copy16(void* lds_dst, const void* gsrc) {
#if defined(__gfx1250__)
  const unsigned int l = (unsigned int)(unsigned long long)lds_dst;  // LDS aperture: addr[31:0]
  asm volatile("global_load_async_to_lds_b128 %0, %1, off"
               :: "v"(l), "v"(gsrc) : "memory");
#else
  *(float4*)lds_dst = *(const float4*)gsrc;  // host-pass placeholder
#endif
}
__device__ __forceinline__ void async_join() {
#if defined(__gfx1250__)
  asm volatile("s_wait_asynccnt 0x0" ::: "memory");
#endif
}

// --------------------------- fp32 -> bf16 convert ---------------------------
__global__ void __launch_bounds__(256) cvt_bf16(const float* __restrict__ in,
                                                __bf16* __restrict__ out, int n4) {
  int i = blockIdx.x * blockDim.x + threadIdx.x;
  const int stride = gridDim.x * blockDim.x;
  for (; i < n4; i += stride) {
    const float4 v = ((const float4*)in)[i];
    v4bf o;
    o[0] = (__bf16)v.x; o[1] = (__bf16)v.y; o[2] = (__bf16)v.z; o[3] = (__bf16)v.w;
    ((v4bf*)out)[i] = o;
  }
}

// --------------------------- input-gate GEMM (bf16 WMMA) --------------------
// Y(M,N) = X(M,K) * W(N,K)^T + (bi+bh)[N], one wave per 16x16 C tile,
// 8 waves per block share the same M tile (A reuse via WGP cache).
__global__ void __launch_bounds__(256)
gate_gemm(const __bf16* __restrict__ X, const __bf16* __restrict__ W,
          const float* __restrict__ bi, const float* __restrict__ bh,
          float* __restrict__ Y, int M, int N, int K) {
  const int lane = threadIdx.x & 31;
  const int wv   = threadIdx.x >> 5;              // 0..7
  const int bpr  = (N >> 4) >> 3;                 // N-tile blocks per M row
  const int tm   = (blockIdx.x / bpr) << 4;
  const int tn   = (((blockIdx.x % bpr) << 3) | wv) << 4;
  const int half = lane >> 4;                     // 0 or 1
  const int ln   = lane & 15;
  // A fragment: M = ln, K chunks {k0..k0+7, k0+16..k0+23}, k0 = 8*half.
  // B fragment mirrors A with N = ln; B col n == row n of W (contiguous K).
  const __bf16* arow = X + (size_t)(tm + ln) * K + half * 8;
  const __bf16* brow = W + (size_t)(tn + ln) * K + half * 8;
  v8f acc = {};
  for (int k = 0; k < K; k += 32) {
    v8bf a0 = *(const v8bf*)(arow + k);
    v8bf a1 = *(const v8bf*)(arow + k + 16);
    v8bf b0 = *(const v8bf*)(brow + k);
    v8bf b1 = *(const v8bf*)(brow + k + 16);
    v16bf a, b;
#pragma unroll
    for (int i = 0; i < 8; ++i) {
      a[i] = a0[i]; a[i + 8] = a1[i];
      b[i] = b0[i]; b[i + 8] = b1[i];
    }
    acc = __builtin_amdgcn_wmma_f32_16x16x32_bf16(
        false, a, false, b, (short)0, acc, false, false);
  }
  const float bias = bi[tn + ln] + bh[tn + ln];
#pragma unroll
  for (int r = 0; r < 8; ++r) {               // C layout: M = r + 8*half, N = ln
    const int m = tm + r + half * 8;
    Y[(size_t)m * N + tn + ln] = acc[r] + bias;
  }
}

// --------------------------- LSTM scan (fp32 WMMA) --------------------------
// 32 WGs x 128 threads. WG owns a 16-hidden-unit chunk; wave w owns gate w.
// w_hh slice (64x512 f32, stride-padded) lives in dynamic LDS for the whole
// scan (320KB/WGP makes this possible); h staged to LDS per step with async
// b128 copies; grid barrier via monotonic atomic counter; h double-buffered
// so one barrier per step suffices. fp32 WMMA keeps recurrence at reference
// precision (errors would compound over 1024 steps in bf16; the scan is
// latency-bound so fp32 math throughput is free).
__global__ void __launch_bounds__(128)
lstm_scan(const float* __restrict__ xg, const float* __restrict__ w_hh,
          float* __restrict__ h0buf, float* __restrict__ h1buf,
          float* __restrict__ out, int col_off, int rev, int* bar) {
  __shared__ float sh_h[16 * 516];
  __shared__ float sh_gate[4][16][16];
  __shared__ float sh_c[16][16];
  extern __shared__ float sh_w[];               // 64 * 516 floats = 132096 B

  const int tid  = threadIdx.x;
  const int wave = tid >> 5;                    // gate group: 0=i 1=f 2=g 3=o
  const int lane = tid & 31;
  const int half = lane >> 4;
  const int ln   = lane & 15;
  const int hc   = blockIdx.x * 16;             // hidden chunk base

  // Stage w_hh slice via async b128: rows n = g*CH + hc + j (contiguous K).
  for (int c = tid; c < 64 * 128; c += 128) {   // 64 rows x 128 16B chunks
    const int rl = c >> 7, k4 = (c & 127) << 2;
    async_copy16(sh_w + rl * 516 + k4,
                 w_hh + (size_t)((rl >> 4) * CH + hc + (rl & 15)) * CH + k4);
  }
  // Zero cell state + this WG's slice of h0 (read at step 0).
  for (int i = tid; i < 256; i += 128) {
    sh_c[i >> 4][i & 15] = 0.f;
    h0buf[(size_t)(i >> 4) * CH + hc + (i & 15)] = 0.f;
  }
  async_join();
  int round = 0;
  // ---- grid barrier (round 0: everyone sees zeroed h0) ----
  __threadfence(); __syncthreads();
  if (tid == 0) __hip_atomic_fetch_add(bar, 1, __ATOMIC_ACQ_REL, __HIP_MEMORY_SCOPE_AGENT);
  ++round;
  if (tid == 0)
    while (__hip_atomic_load(bar, __ATOMIC_ACQUIRE, __HIP_MEMORY_SCOPE_AGENT) < round * NWG)
      __builtin_amdgcn_s_sleep(1);
  __syncthreads();

  for (int s = 0; s < CT; ++s) {
    const int t = rev ? (CT - 1 - s) : s;
    const float* hin  = (s & 1) ? h1buf : h0buf;
    float*       hout = (s & 1) ? h0buf : h1buf;

    // Stage full h (16 x 512 f32 = 32KB) into LDS (padded stride) async.
    for (int c = tid; c < 2048; c += 128) {     // 2048 16B chunks
      const int row = c >> 7, colf = (c & 127) << 2;
      async_copy16(sh_h + row * 516 + colf, hin + (size_t)row * CH + colf);
    }
    async_join();
    __syncthreads();

    // 16x16 gate tile = h(16xK) @ w_sliceᵀ, K=512, fp32 WMMA K=4 steps.
    v8f acc = {};
    const float* abase = sh_h + ln * 516 + half * 2;                // A: M=ln
    const float* bbase = sh_w + (wave * 16 + ln) * 516 + half * 2;  // B: N=ln
#pragma unroll 4
    for (int k = 0; k < CH; k += 4) {
      v2f av = *(const v2f*)(abase + k);
      v2f bv = *(const v2f*)(bbase + k);
      acc = __builtin_amdgcn_wmma_f32_16x16x4_f32(
          false, av, false, bv, (short)0, acc, false, false);
    }
    // Add precomputed input gates, apply nonlinearity (uniform per wave).
    const int gcol = wave * CH + hc + ln;
#pragma unroll
    for (int r = 0; r < 8; ++r) {
      const int m = r + half * 8;
      float g = acc[r] + xg[((size_t)m * CT + t) * C4H + gcol];
      g = (wave == 2) ? tanhf(g) : (1.f / (1.f + __expf(-g)));
      sh_gate[wave][m][ln] = g;
    }
    __syncthreads();

    // c = f*c + i*g ; h = o*tanh(c); write h chunk + layer output.
    for (int i = tid; i < 256; i += 128) {
      const int m = i >> 4, n = i & 15;
      const float c = sh_gate[1][m][n] * sh_c[m][n] +
                      sh_gate[0][m][n] * sh_gate[2][m][n];
      sh_c[m][n] = c;
      const float h = sh_gate[3][m][n] * tanhf(c);
      hout[(size_t)m * CH + hc + n] = h;
      out[((size_t)m * CT + t) * C2H + col_off + hc + n] = h;
    }
    // ---- grid barrier (release h writes to all WGs) ----
    __threadfence(); __syncthreads();
    if (tid == 0) __hip_atomic_fetch_add(bar, 1, __ATOMIC_ACQ_REL, __HIP_MEMORY_SCOPE_AGENT);
    ++round;
    if (tid == 0)
      while (__hip_atomic_load(bar, __ATOMIC_ACQUIRE, __HIP_MEMORY_SCOPE_AGENT) < round * NWG)
        __builtin_amdgcn_s_sleep(1);
    __syncthreads();
  }
}

// --------------------------- attention pooling ------------------------------
__global__ void __launch_bounds__(256)
attn_pool(const float* __restrict__ seq, const float* __restrict__ aw,
          const float* __restrict__ ab, float* __restrict__ enc) {
  __shared__ float sw[C2H];
  __shared__ float sc[CT];
  __shared__ float red[256];
  const int b = blockIdx.x, tid = threadIdx.x;
  const float* base = seq + (size_t)b * CT * C2H;
  for (int i = tid; i < C2H; i += 256) sw[i] = aw[i];
  __syncthreads();
  for (int t = tid; t < CT; t += 256) {
    const float* row = base + (size_t)t * C2H;
    float s = 0.f;
    for (int h = 0; h < C2H; h += 4) {
      float4 v = *(const float4*)(row + h);
      s += v.x * sw[h] + v.y * sw[h + 1] + v.z * sw[h + 2] + v.w * sw[h + 3];
    }
    sc[t] = s + ab[0];
  }
  __syncthreads();
  float m = -3.4e38f;
  for (int t = tid; t < CT; t += 256) m = fmaxf(m, sc[t]);
  red[tid] = m; __syncthreads();
  for (int off = 128; off > 0; off >>= 1) {
    if (tid < off) red[tid] = fmaxf(red[tid], red[tid + off]);
    __syncthreads();
  }
  const float mx = red[0];
  __syncthreads();
  float sum = 0.f;
  for (int t = tid; t < CT; t += 256) { float e = __expf(sc[t] - mx); sc[t] = e; sum += e; }
  red[tid] = sum; __syncthreads();
  for (int off = 128; off > 0; off >>= 1) {
    if (tid < off) red[tid] += red[tid + off];
    __syncthreads();
  }
  const float inv = 1.f / red[0];
  __syncthreads();
  for (int h = tid; h < C2H; h += 256) {
    float acc = 0.f;
    for (int t = 0; t < CT; ++t) acc += sc[t] * base[(size_t)t * C2H + h];
    enc[(size_t)b * C2H + h] = acc * inv;
  }
}

// --------------------------- host orchestration -----------------------------
extern "C" void kernel_launch(void* const* d_in, const int* in_sizes, int n_in,
                              void* d_out, int out_size, void* d_ws, size_t ws_size,
                              hipStream_t stream) {
  (void)in_sizes; (void)n_in; (void)out_size; (void)ws_size;
  const float* x         = (const float*)d_in[0];
  const float* w_ih_l0   = (const float*)d_in[1];
  const float* w_hh_l0   = (const float*)d_in[2];
  const float* b_ih_l0   = (const float*)d_in[3];
  const float* b_hh_l0   = (const float*)d_in[4];
  const float* w_ih_l0r  = (const float*)d_in[5];
  const float* w_hh_l0r  = (const float*)d_in[6];
  const float* b_ih_l0r  = (const float*)d_in[7];
  const float* b_hh_l0r  = (const float*)d_in[8];
  const float* w_ih_l1   = (const float*)d_in[9];
  const float* w_hh_l1   = (const float*)d_in[10];
  const float* b_ih_l1   = (const float*)d_in[11];
  const float* b_hh_l1   = (const float*)d_in[12];
  const float* w_ih_l1r  = (const float*)d_in[13];
  const float* w_hh_l1r  = (const float*)d_in[14];
  const float* b_ih_l1r  = (const float*)d_in[15];
  const float* b_hh_l1r  = (const float*)d_in[16];
  const float* attn_w    = (const float*)d_in[17];
  const float* attn_b    = (const float*)d_in[18];
  float* enc = (float*)d_out;

  // Workspace carve-up.
  char* ws = (char*)d_ws;
  size_t off = 0;
  int*    bars  = (int*)(ws + off);     off += 256;
  __bf16* xbf   = (__bf16*)(ws + off);  off += (size_t)CB * CT * 2048 * 2;  // 64 MiB slot
  __bf16* wbf_f = (__bf16*)(ws + off);  off += (size_t)C4H * 2048 * 2;      // 8 MiB
  __bf16* wbf_b = (__bf16*)(ws + off);  off += (size_t)C4H * 2048 * 2;
  float*  xg_f  = (float*)(ws + off);   off += (size_t)CB * CT * C4H * 4;   // 128 MiB
  float*  xg_b  = (float*)(ws + off);   off += (size_t)CB * CT * C4H * 4;
  float*  out0  = (float*)(ws + off);   off += (size_t)CB * CT * C2H * 4;   // 64 MiB
  float*  out1  = (float*)(ws + off);   off += (size_t)CB * CT * C2H * 4;
  float*  h0    = (float*)(ws + off);   off += (size_t)CB * CH * 4;
  float*  h1    = (float*)(ws + off);   off += (size_t)CB * CH * 4;

  const int scan_lds = 64 * 516 * 4;    // dynamic LDS for w_hh slice (129 KiB)
  hipFuncSetAttribute((const void*)lstm_scan,
                      hipFuncAttributeMaxDynamicSharedMemorySize, 160 * 1024);
  hipMemsetAsync(bars, 0, 256, stream);

  const int M = CB * CT;                                // 16384
  const int gemm_blocks = (M / 16) * ((C4H / 16) / 8);  // 16384

  // ---------------- layer 0 ----------------
  cvt_bf16<<<2048, 256, 0, stream>>>(x, xbf, CB * CT * 2048 / 4);
  cvt_bf16<<<1024, 256, 0, stream>>>(w_ih_l0,  wbf_f, C4H * 2048 / 4);
  cvt_bf16<<<1024, 256, 0, stream>>>(w_ih_l0r, wbf_b, C4H * 2048 / 4);
  gate_gemm<<<gemm_blocks, 256, 0, stream>>>(xbf, wbf_f, b_ih_l0,  b_hh_l0,  xg_f, M, C4H, 2048);
  gate_gemm<<<gemm_blocks, 256, 0, stream>>>(xbf, wbf_b, b_ih_l0r, b_hh_l0r, xg_b, M, C4H, 2048);
  lstm_scan<<<NWG, 128, scan_lds, stream>>>(xg_f, w_hh_l0,  h0, h1, out0, 0,  0, bars + 0);
  lstm_scan<<<NWG, 128, scan_lds, stream>>>(xg_b, w_hh_l0r, h0, h1, out0, CH, 1, bars + 1);

  // ---------------- layer 1 ----------------
  cvt_bf16<<<2048, 256, 0, stream>>>(out0, xbf, CB * CT * C2H / 4);
  cvt_bf16<<<1024, 256, 0, stream>>>(w_ih_l1,  wbf_f, C4H * C2H / 4);
  cvt_bf16<<<1024, 256, 0, stream>>>(w_ih_l1r, wbf_b, C4H * C2H / 4);
  gate_gemm<<<gemm_blocks, 256, 0, stream>>>(xbf, wbf_f, b_ih_l1,  b_hh_l1,  xg_f, M, C4H, C2H);
  gate_gemm<<<gemm_blocks, 256, 0, stream>>>(xbf, wbf_b, b_ih_l1r, b_hh_l1r, xg_b, M, C4H, C2H);
  lstm_scan<<<NWG, 128, scan_lds, stream>>>(xg_f, w_hh_l1,  h0, h1, out1, 0,  0, bars + 2);
  lstm_scan<<<NWG, 128, scan_lds, stream>>>(xg_b, w_hh_l1r, h0, h1, out1, CH, 1, bars + 3);

  // ---------------- attention pooling ----------------
  attn_pool<<<CB, 256, 0, stream>>>(out1, attn_w, attn_b, enc);
}
